// ACE_NODE_74208444940289
// MI455X (gfx1250) — matrix operand order repaired and checked by reading, not compiled, not run on hardware
//
#include <hip/hip_runtime.h>

typedef __attribute__((ext_vector_type(16))) _Float16 v16h;
typedef __attribute__((ext_vector_type(8)))  _Float16 v8h;
typedef __attribute__((ext_vector_type(8)))  float    v8f;
typedef __attribute__((ext_vector_type(4)))  float    v4f;

#define B_   256
#define S_   512
#define DI_  64
#define H_   128
#define W_   64
#define NSUB 2

// ---------------- WMMA helpers (CDNA5 wave32, 16x16x32 f16 -> f32) ----------------

__device__ __forceinline__ v8f vzero() {
  v8f z = {0.f,0.f,0.f,0.f,0.f,0.f,0.f,0.f};
  return z;
}

__device__ __forceinline__ v8f wmma16(v16h a, v16h b, v8f c) {
  // (neg_a, A, neg_b, B, c_mod, C, reuse_a, reuse_b)
  return __builtin_amdgcn_wmma_f32_16x16x32_f16(false, a, false, b, (short)0, c, false, false);
}

// A fragment: 16x32 f16 tile of activation matrix (row-major, leading dim lda).
// ISA layout: lane L (m = L&15, hi = L>>4): elems 0..7 = K(hi*8 + 0..7),
//             elems 8..15 = K(16 + hi*8 + 0..7).
__device__ __forceinline__ v16h load_a(const _Float16* A, int lda, int kbase, int lane) {
  const int m  = lane & 15;
  const int hi = (lane >> 4) & 1;
  const _Float16* p = A + m * lda + kbase + hi * 8;
  v8h lo = *(const v8h*)(p);
  v8h hh = *(const v8h*)(p + 16);
  v16h a;
#pragma unroll
  for (int e = 0; e < 8; ++e) { a[e] = lo[e]; a[8 + e] = hh[e]; }
  return a;
}

// B fragment: 32x16 slice of W^T, with W stored row-major (N_out x K_in, leading dim ldw).
// B[k][n] = W[n][k]. Lane L: n = L&15, hi = L>>4: elems 0..7 = K(hi*8 + 0..7), elems 8..15 = K(16 + hi*8 + 0..7).
__device__ __forceinline__ v16h load_b(const _Float16* Wt, int ldw, int kbase, int nbase, int lane) {
  const int n    = lane & 15;
  const int koff = (lane >> 4) * 8;
  const _Float16* p = Wt + (size_t)(nbase + n) * ldw + kbase + koff;
  v8h lo = *(const v8h*)(p);
  v8h hh = *(const v8h*)(p + 16);
  v16h b;
#pragma unroll
  for (int e = 0; e < 8; ++e) { b[e] = lo[e]; b[8 + e] = hh[e]; }
  return b;
}

// GEMM tile with TWO independent accumulation chains (even/odd k) so a single
// wave keeps >=2 WMMAs in flight in the XDL pipe (we run 1 wave/SIMD).
template <int K>
__device__ __forceinline__ void gemm_acc(v8f& acc, const _Float16* A, int lda,
                                         const _Float16* Wt, int ldw, int nbase, int lane) {
  if constexpr (K <= 32) {
    acc = wmma16(load_a(A, lda, 0, lane), load_b(Wt, ldw, 0, nbase, lane), acc);
  } else {
    v8f c1 = vzero();
#pragma unroll
    for (int k = 0; k < K; k += 64) {
      acc = wmma16(load_a(A, lda, k, lane), load_b(Wt, ldw, k, nbase, lane), acc);
      c1  = wmma16(load_a(A, lda, k + 32, lane), load_b(Wt, ldw, k + 32, nbase, lane), c1);
    }
#pragma unroll
    for (int r = 0; r < 8; ++r) acc[r] += c1[r];
  }
}

// Store a C/D fragment (f32) as f16 into an LDS row-major buffer.
// C/D layout: lane L: n = L&15, rows m = (L>>4)*8 + r for r in 0..7.
__device__ __forceinline__ void store_cd(_Float16* buf, int ld, int nbase, int lane, v8f v) {
  const int n  = lane & 15;
  const int mb = (lane >> 4) * 8;
#pragma unroll
  for (int r = 0; r < 8; ++r) buf[(mb + r) * ld + nbase + n] = (_Float16)v[r];
}

__device__ __forceinline__ float softplus(float x) {
  return x > 20.f ? x : __logf(1.f + __expf(x));
}
__device__ __forceinline__ float sigm(float x) { return 1.f / (1.f + __expf(-x)); }

// ---- Runge-Kutta slope storage in LDS (lane-private, no barriers needed) ----
// Block of 256 floats per (j, wave, g): [half(2)][lane(32)][4 floats] so each
// ds_load_b128/ds_store_b128 has consecutive lanes on consecutive 16B units.
__device__ __forceinline__ void kk_store(float* kbuf, int j, int wv, int g, int lane, v8f v) {
  float* p = kbuf + (((j * 4 + wv) * 2 + g) << 8) + lane * 4;
  v4f lo = {v[0], v[1], v[2], v[3]};
  v4f hi = {v[4], v[5], v[6], v[7]};
  *(v4f*)p = lo;
  *(v4f*)(p + 128) = hi;
}
__device__ __forceinline__ v8f kk_load(const float* kbuf, int j, int wv, int g, int lane) {
  const float* p = kbuf + (((j * 4 + wv) * 2 + g) << 8) + lane * 4;
  v4f lo = *(const v4f*)p;
  v4f hi = *(const v4f*)(p + 128);
  v8f v = {lo[0], lo[1], lo[2], lo[3], hi[0], hi[1], hi[2], hi[3]};
  return v;
}

// z := y + dt * sum_j cf[j] * k_j  for this wave's columns, written to sZ as f16.
template <int NK>
__device__ __forceinline__ void put_z(_Float16* sZ, const float* kbuf, const v8f y[2],
                                      const float* cf, float dt, int n0, int wv, int lane) {
#pragma unroll
  for (int g = 0; g < 2; ++g) {
    v8f z = y[g];
#pragma unroll
    for (int j = 0; j < NK; ++j) {
      const v8f kv = kk_load(kbuf, j, wv, g, lane);
      const float s = dt * cf[j];
#pragma unroll
      for (int r = 0; r < 8; ++r) z[r] += s * kv[r];
    }
    store_cd(sZ, H_, n0 + 64 * g, lane, z);
  }
}

// One vector-field evaluation: k_j = MLP([y_stage, t]); result -> kbuf slot j.
__device__ __forceinline__ void mlp_eval(float tcur, int j, int n0, int wv, int lane,
                                         float* kbuf,
                                         const _Float16* sZ, _Float16* sH0, _Float16* sH1,
                                         const _Float16* sW0h, const _Float16* sW1,
                                         const _Float16* sW2, const _Float16* sW3,
                                         const float* sB0, const float* sW0t,
                                         const float* sB1, const float* sB2, const float* sB3) {
  const int n = lane & 15;
  __syncthreads();                         // z writes visible
  // L0: (16xH) @ W0h^T -> 16xW, softplus; time column folded into bias
  {
    v8f a0 = vzero();
    gemm_acc<H_>(a0, sZ, H_, sW0h, H_, n0, lane);
    const float be = sB0[n0 + n] + tcur * sW0t[n0 + n];
    v8f h;
#pragma unroll
    for (int r = 0; r < 8; ++r) h[r] = softplus(a0[r] + be);
    store_cd(sH0, W_, n0, lane, h);
  }
  __syncthreads();
  // L1
  {
    v8f a1 = vzero();
    gemm_acc<W_>(a1, sH0, W_, sW1, W_, n0, lane);
    const float be = sB1[n0 + n];
    v8f h;
#pragma unroll
    for (int r = 0; r < 8; ++r) h[r] = softplus(a1[r] + be);
    store_cd(sH1, W_, n0, lane, h);
  }
  __syncthreads();
  // L2 (reuse sH0)
  {
    v8f a2 = vzero();
    gemm_acc<W_>(a2, sH1, W_, sW2, W_, n0, lane);
    const float be = sB2[n0 + n];
    v8f h;
#pragma unroll
    for (int r = 0; r < 8; ++r) h[r] = softplus(a2[r] + be);
    store_cd(sH0, W_, n0, lane, h);
  }
  __syncthreads();
  // L3: (16xW) @ W3^T -> 16xH, identity; write straight to the k_j LDS slot
#pragma unroll
  for (int g = 0; g < 2; ++g) {
    v8f a3 = vzero();
    gemm_acc<W_>(a3, sH0, W_, sW3, W_, n0 + 64 * g, lane);
    const float be = sB3[n0 + 64 * g + n];
#pragma unroll
    for (int r = 0; r < 8; ++r) a3[r] += be;
    kk_store(kbuf, j, wv, g, lane, a3);
  }
}

// ---------------- prologue: fp32 -> fp16 GRU weights into workspace ----------------

__global__ void cvt_gru_weights(const float* __restrict__ wih, const float* __restrict__ whh,
                                _Float16* __restrict__ ws) {
  int i = blockIdx.x * 256 + threadIdx.x;
  if (i < 3 * H_ * DI_) ws[i] = (_Float16)wih[i];
  if (i < 3 * H_ * H_)  ws[3 * H_ * DI_ + i] = (_Float16)whh[i];
}

// ---------------- persistent scan kernel: 16 batch rows per workgroup ----------------

__global__ void __launch_bounds__(128, 1)
ace_node_scan(const float* __restrict__ x_seq, const float* __restrict__ y0,
              const float* __restrict__ W0, const float* __restrict__ b0,
              const float* __restrict__ W1, const float* __restrict__ b1,
              const float* __restrict__ W2, const float* __restrict__ b2,
              const float* __restrict__ W3, const float* __restrict__ b3,
              const _Float16* __restrict__ gWih, const _Float16* __restrict__ gWhh,
              const float* __restrict__ gru_b, const float* __restrict__ gru_bn,
              float* __restrict__ out) {
  __shared__ _Float16 sW0h[W_ * H_];        // 16 KB
  __shared__ _Float16 sW1[W_ * W_];         //  8 KB
  __shared__ _Float16 sW2[W_ * W_];         //  8 KB
  __shared__ _Float16 sW3[H_ * W_];         // 16 KB
  __shared__ _Float16 sZ[16 * H_];          //  4 KB  stage input / h for GRU
  __shared__ _Float16 sH0[16 * W_];         //  2 KB
  __shared__ _Float16 sH1[16 * W_];         //  2 KB
  __shared__ _Float16 sX[16 * DI_];         //  2 KB
  __shared__ float    sKK[6 * 4 * 2 * 256]; // 48 KB: RK slopes k1..k6 (lane-private)
  __shared__ float sB0[W_], sW0t[W_], sB1[W_], sB2[W_], sB3[H_];
  __shared__ float sGB[3 * H_], sGBN[H_];

  const int tid  = threadIdx.x;
  const int lane = tid & 31;
  const int wv   = tid >> 5;
  const int n0   = wv * 16;              // wave's first N column
  const int b0r  = blockIdx.x * 16;      // batch-row base

  // ---- load + convert MLP weights into LDS (once; reused 512x12 times) ----
  for (int i = tid; i < W_ * H_; i += 128) sW0h[i] = (_Float16)W0[(i >> 7) * (H_ + 1) + (i & 127)];
  for (int i = tid; i < W_ * W_; i += 128) { sW1[i] = (_Float16)W1[i]; sW2[i] = (_Float16)W2[i]; }
  for (int i = tid; i < H_ * W_; i += 128) sW3[i] = (_Float16)W3[i];
  for (int i = tid; i < W_; i += 128) {
    sB0[i] = b0[i]; sW0t[i] = W0[i * (H_ + 1) + H_]; sB1[i] = b1[i]; sB2[i] = b2[i];
  }
  for (int i = tid; i < H_; i += 128) { sB3[i] = b3[i]; sGBN[i] = gru_bn[i]; }
  for (int i = tid; i < 3 * H_; i += 128) sGB[i] = gru_b[i];

  // ---- y state in registers, WMMA C/D layout: y[g] covers columns n0+64g..+15 ----
  v8f y[2];
  {
    const int n = lane & 15, mb = (lane >> 4) * 8;
#pragma unroll
    for (int g = 0; g < 2; ++g)
#pragma unroll
      for (int r = 0; r < 8; ++r)
        y[g][r] = y0[(size_t)(b0r + mb + r) * H_ + n0 + 64 * g + n];
  }
  __syncthreads();

  const float dt = 1.0f / NSUB;
  // Tsit5 tableau
  const float C2 = 0.161f, C3 = 0.327f, C4 = 0.9f, C5 = 0.9800255409045097f, C6 = 1.0f;
  const float A21 = 0.161f;
  const float A31 = -0.008480655492356989f, A32 = 0.335480655492357f;
  const float A41 = 2.8971530571054935f, A42 = -6.359448489975075f, A43 = 4.3622954328695815f;
  const float A51 = 5.325864828439257f, A52 = -11.748883564062828f, A53 = 7.4955393428898365f,
              A54 = -0.09249506636175525f;
  const float A61 = 5.86145544294642f, A62 = -12.92096931784711f, A63 = 8.159367898576159f,
              A64 = -0.071584973281401f, A65 = -0.028269050394068383f;
  const float WB1 = 0.09646076681806523f, WB2 = 0.01f, WB3 = 0.4798896504144996f,
              WB4 = 1.379008574103742f, WB5 = -3.290069515436081f, WB6 = 2.324710524099774f;

  // x-row assignment for the register-staged GRU input: 16 rows x 64 cols / 128 thr
  const int xrow = tid >> 3;            // 0..15
  const int xcol = (tid & 7) * 8;       // 0,8,..,56

#pragma unroll 1
  for (int t = 0; t < S_; ++t) {
    const float tstep = (float)t;

    // Issue x_t global loads NOW; they complete during the 12 MLP stage evals,
    // so the GRU never waits on L2/HBM.
    float xr[8];
    {
      const float* xp = x_seq + ((size_t)(b0r + xrow) * S_ + t) * DI_ + xcol;
#pragma unroll
      for (int e = 0; e < 8; ++e) xr[e] = xp[e];
      if (t + 1 < S_ && tid < 16) {
        // warm next step's rows in L2/WGP$ (global_prefetch)
        __builtin_prefetch(x_seq + ((size_t)(b0r + tid) * S_ + (t + 1)) * DI_, 0, 1);
      }
    }

#pragma unroll 1
    for (int sub = 0; sub < NSUB; ++sub) {
      const float tb = tstep + (float)sub * dt;
      // k1
      put_z<0>(sZ, sKK, y, nullptr, dt, n0, wv, lane);
      mlp_eval(tb, 0, n0, wv, lane, sKK, sZ, sH0, sH1, sW0h, sW1, sW2, sW3,
               sB0, sW0t, sB1, sB2, sB3);
      // k2
      { const float cf[1] = {A21};
        put_z<1>(sZ, sKK, y, cf, dt, n0, wv, lane); }
      mlp_eval(tb + C2 * dt, 1, n0, wv, lane, sKK, sZ, sH0, sH1, sW0h, sW1, sW2, sW3,
               sB0, sW0t, sB1, sB2, sB3);
      // k3
      { const float cf[2] = {A31, A32};
        put_z<2>(sZ, sKK, y, cf, dt, n0, wv, lane); }
      mlp_eval(tb + C3 * dt, 2, n0, wv, lane, sKK, sZ, sH0, sH1, sW0h, sW1, sW2, sW3,
               sB0, sW0t, sB1, sB2, sB3);
      // k4
      { const float cf[3] = {A41, A42, A43};
        put_z<3>(sZ, sKK, y, cf, dt, n0, wv, lane); }
      mlp_eval(tb + C4 * dt, 3, n0, wv, lane, sKK, sZ, sH0, sH1, sW0h, sW1, sW2, sW3,
               sB0, sW0t, sB1, sB2, sB3);
      // k5
      { const float cf[4] = {A51, A52, A53, A54};
        put_z<4>(sZ, sKK, y, cf, dt, n0, wv, lane); }
      mlp_eval(tb + C5 * dt, 4, n0, wv, lane, sKK, sZ, sH0, sH1, sW0h, sW1, sW2, sW3,
               sB0, sW0t, sB1, sB2, sB3);
      // k6
      { const float cf[5] = {A61, A62, A63, A64, A65};
        put_z<5>(sZ, sKK, y, cf, dt, n0, wv, lane); }
      mlp_eval(tb + C6 * dt, 5, n0, wv, lane, sKK, sZ, sH0, sH1, sW0h, sW1, sW2, sW3,
               sB0, sW0t, sB1, sB2, sB3);
      // 5th-order update: y += dt * sum_j wb[j] * k_j
      {
        const float wf[6] = {WB1, WB2, WB3, WB4, WB5, WB6};
#pragma unroll
        for (int g = 0; g < 2; ++g)
#pragma unroll
          for (int j = 0; j < 6; ++j) {
            const v8f kv = kk_load(sKK, j, wv, g, lane);
            const float s = dt * wf[j];
#pragma unroll
            for (int r = 0; r < 8; ++r) y[g][r] += s * kv[r];
          }
      }
    }

    // ---- GRU cell ----
    {
      // stage pre-loaded x_t into LDS (one packed 16B store per thread)
      v8h xh;
#pragma unroll
      for (int e = 0; e < 8; ++e) xh[e] = (_Float16)xr[e];
      *(v8h*)(sX + xrow * DI_ + xcol) = xh;
      // stage h=y into sZ (f16) for the whh GEMM
#pragma unroll
      for (int g = 0; g < 2; ++g) store_cd(sZ, H_, n0 + 64 * g, lane, y[g]);
    }
    __syncthreads();

    // process one column group at a time to keep gate accumulators' live range short
#pragma unroll
    for (int g = 0; g < 2; ++g) {
      const int c0 = n0 + 64 * g;
      v8f ra = vzero();
      gemm_acc<DI_>(ra, sX, DI_, gWih, DI_, c0, lane);            // ir
      gemm_acc<H_>(ra, sZ, H_, gWhh, H_, c0, lane);               // + hr
      v8f za = vzero();
      gemm_acc<DI_>(za, sX, DI_, gWih, DI_, H_ + c0, lane);       // iz
      gemm_acc<H_>(za, sZ, H_, gWhh, H_, H_ + c0, lane);          // + hz
      v8f na = vzero();
      gemm_acc<DI_>(na, sX, DI_, gWih, DI_, 2 * H_ + c0, lane);   // ig_n
      v8f ha = vzero();
      gemm_acc<H_>(ha, sZ, H_, gWhh, H_, 2 * H_ + c0, lane);      // hn

      const int n = lane & 15, mb = (lane >> 4) * 8;
      const int c = c0 + n;
      const float gbR = sGB[c], gbZ = sGB[H_ + c], gbN = sGB[2 * H_ + c], gbnv = sGBN[c];
#pragma unroll
      for (int r = 0; r < 8; ++r) {
        const float rr = sigm(ra[r] + gbR);
        const float zz = sigm(za[r] + gbZ);
        const float nv = tanhf(na[r] + gbN + rr * (ha[r] + gbnv));
        const float hv = y[g][r];
        const float o  = nv + zz * (hv - nv);
        y[g][r] = o;
        out[((size_t)(b0r + mb + r) * S_ + t) * H_ + c] = o;
      }
    }
    __syncthreads();   // protect sX/sZ before next step rewrites them
  }
}

extern "C" void kernel_launch(void* const* d_in, const int* in_sizes, int n_in,
                              void* d_out, int out_size, void* d_ws, size_t ws_size,
                              hipStream_t stream) {
  const float* x_seq = (const float*)d_in[0];
  const float* y0v   = (const float*)d_in[1];
  // d_in[2] = attn (unused by the module)
  const float* W0v = (const float*)d_in[3];
  const float* b0v = (const float*)d_in[4];
  const float* W1v = (const float*)d_in[5];
  const float* b1v = (const float*)d_in[6];
  const float* W2v = (const float*)d_in[7];
  const float* b2v = (const float*)d_in[8];
  const float* W3v = (const float*)d_in[9];
  const float* b3v = (const float*)d_in[10];
  const float* gwih = (const float*)d_in[11];
  const float* gwhh = (const float*)d_in[12];
  const float* gbv  = (const float*)d_in[13];
  const float* gbnv = (const float*)d_in[14];
  float* out = (float*)d_out;
  _Float16* wsH = (_Float16*)d_ws;   // [3H*DI] wih f16, then [3H*H] whh f16

  cvt_gru_weights<<<(3 * H_ * H_ + 255) / 256, 256, 0, stream>>>(gwih, gwhh, wsH);
  ace_node_scan<<<B_ / 16, 128, 0, stream>>>(x_seq, y0v, W0v, b0v, W1v, b1v, W2v, b2v,
                                             W3v, b3v, wsH, wsH + 3 * H_ * DI_,
                                             gbv, gbnv, out);
}
